// KernelFilter_S_55070070670036
// MI455X (gfx1250) — matrix operand order repaired and checked by reading, not compiled, not run on hardware
//
#include <hip/hip_runtime.h>

typedef __attribute__((ext_vector_type(16))) __bf16        v16bf;
typedef __attribute__((ext_vector_type(8)))  float         v8f;
typedef __attribute__((ext_vector_type(4)))  unsigned int  v4u;

union U32x4 { v4u q[2]; v16bf v; };

// Padded NHWC geometry: [B][66][66][IC], logical (h,w) at padded (h+1,w+1).
#define PH 66

// ---------------------------------------------------------------------------
// Generic 32-byte fragment load: two b128s at element offsets from a
// precomputed per-lane base pointer (lowers to immediate-offset loads after
// strength reduction). step = 8 for both A and B fragments.
// ---------------------------------------------------------------------------
__device__ inline v16bf load_frag(const __bf16* __restrict__ base, int off, int step) {
  U32x4 f;
  f.q[0] = *reinterpret_cast<const v4u*>(base + off);
  f.q[1] = *reinterpret_cast<const v4u*>(base + off + step);
  return f.v;
}

// C/D 16x16 f32: VGPR r, lane l -> (M = r + 8*(l>=16), N = l&15)
template <int OC, bool LEAKY, bool F32OUT>
__device__ inline void store_out(v8f acc, const float* __restrict__ bias,
                                 __bf16* __restrict__ Ybf, float* __restrict__ Yf,
                                 const float* __restrict__ resid,
                                 int b, int h, int w, int mtile, int hi) {
  if constexpr (!F32OUT) {
    union { v4u q; __bf16 e[8]; } o;
#pragma unroll
    for (int r = 0; r < 8; ++r) {
      int oc = mtile * 16 + hi * 8 + r;
      float v = acc[r] + (bias ? bias[oc] : 0.0f);
      if constexpr (LEAKY) v = (v >= 0.0f) ? v : 0.2f * v;
      o.e[r] = (__bf16)v;
    }
    size_t pix = ((size_t)b * PH + (h + 1)) * PH + (w + 1);
    *reinterpret_cast<v4u*>(Ybf + pix * OC + mtile * 16 + hi * 8) = o.q;
  } else {
#pragma unroll
    for (int r = 0; r < 8; ++r) {
      int oc = mtile * 16 + hi * 8 + r;
      float v = acc[r] + bias[oc];
      size_t idx = (((size_t)b * OC + oc) * 64 + h) * 64 + w;
      Yf[idx] = v + resid[idx];
    }
  }
}

// ---------------------------------------------------------------------------
// Implicit-GEMM 3x3 SAME conv via v_wmma_f32_16x16x32_bf16, 2 rows per wave.
// X: padded [B,66,66,IC] bf16.  A: [tap][OC][IC] bf16 (optionally per-batch).
// Block = 256 threads = 8 waves; wave -> (mgroup = wave>>2, ntile = wave&3).
// Fragment addresses = per-lane base register + immediate offset; the K-chunk
// and m-tile loops are unrolled x2 (not fully) so the register allocator can
// keep many fragment buffers in flight (deep clauses + partial loadcnt waits).
// Per chunk: 12 B + 9 A fragment loads feed 18 wmmas (A shared by both rows).
// Fragment layouts used here (same contiguous K order for A and B):
//   B 32x16: lane&15 = column N, lanes<16 hold K=0..15, lanes>=16 K=16..31.
//   A 16x32: lane&15 = row M, lanes<16 K=0..15, lanes>=16 K=16..31.
// ---------------------------------------------------------------------------
template <int IC, int OC, bool LEAKY, bool F32OUT>
__global__ __launch_bounds__(256) void conv3x3_wmma(
    const __bf16* __restrict__ X, const __bf16* __restrict__ A, int perBatchA,
    const float* __restrict__ bias, __bf16* __restrict__ Ybf,
    float* __restrict__ Yf, const float* __restrict__ resid) {
  constexpr int MT = OC / 16;
  constexpr int MPW = MT / 2;  // m-tiles per wave (2 wave-groups of 4)
  static_assert(IC == 32 || MPW == 1, "multi-mtile path requires IC==32");

  int b = blockIdx.y, h0 = blockIdx.x * 2;
  int wave = threadIdx.x >> 5, lane = threadIdx.x & 31;
  int xbase = (wave & 3) * 16;
  int mbase = (wave >> 2) * MPW;
  const __bf16* Xb = X + (size_t)b * PH * PH * IC;
  const __bf16* Ab = A + (perBatchA ? (size_t)b * 9 * OC * IC : (size_t)0);
  int nn = lane & 15, hi = lane >> 4;
  int w = xbase + nn;

  // Per-lane base pointers; fragment loads are immediate offsets from these.
  const __bf16* bbase = Xb + (size_t)(h0 * PH + xbase + nn) * IC + hi * 16;
  const __bf16* abase = Ab + (size_t)(mbase * 16 + nn) * IC + hi * 16;
  // Padded input rows h0..h0+3 cover logical rows h0-1..h0+2.

  if constexpr (IC == 32) {
    v16bf bfr[12];
#pragma unroll
    for (int r = 0; r < 4; ++r)
#pragma unroll
      for (int d = 0; d < 3; ++d)
        bfr[r * 3 + d] = load_frag(bbase, (r * PH + d) * IC, 8);
#pragma unroll 2
    for (int mi = 0; mi < MPW; ++mi) {
      v16bf afr[9];
#pragma unroll
      for (int tap = 0; tap < 9; ++tap)
        afr[tap] = load_frag(abase, (tap * OC + mi * 16) * IC, 8);
      v8f acc0 = {}, acc1 = {};
#pragma unroll
      for (int tap = 0; tap < 9; ++tap) {
        acc0 = __builtin_amdgcn_wmma_f32_16x16x32_bf16(
            false, afr[tap], false, bfr[tap], (short)0, acc0, false, false);
        acc1 = __builtin_amdgcn_wmma_f32_16x16x32_bf16(
            false, afr[tap], false, bfr[tap + 3], (short)0, acc1, false, false);
      }
      store_out<OC, LEAKY, F32OUT>(acc0, bias, Ybf, Yf, resid, b, h0, w, mbase + mi, hi);
      store_out<OC, LEAKY, F32OUT>(acc1, bias, Ybf, Yf, resid, b, h0 + 1, w, mbase + mi, hi);
    }
  } else {
    v8f acc0 = {}, acc1 = {};
#pragma unroll 2
    for (int c0i = 0; c0i < IC / 32; ++c0i) {
      const int c0 = c0i * 32;
      v16bf bfr[12];
#pragma unroll
      for (int r = 0; r < 4; ++r)
#pragma unroll
        for (int d = 0; d < 3; ++d)
          bfr[r * 3 + d] = load_frag(bbase, (r * PH + d) * IC + c0, 8);
      v16bf afr[9];
#pragma unroll
      for (int tap = 0; tap < 9; ++tap)
        afr[tap] = load_frag(abase, tap * OC * IC + c0, 8);
#pragma unroll
      for (int tap = 0; tap < 9; ++tap) {
        acc0 = __builtin_amdgcn_wmma_f32_16x16x32_bf16(
            false, afr[tap], false, bfr[tap], (short)0, acc0, false, false);
        acc1 = __builtin_amdgcn_wmma_f32_16x16x32_bf16(
            false, afr[tap], false, bfr[tap + 3], (short)0, acc1, false, false);
      }
    }
    store_out<OC, LEAKY, F32OUT>(acc0, bias, Ybf, Yf, resid, b, h0, w, mbase, hi);
    store_out<OC, LEAKY, F32OUT>(acc1, bias, Ybf, Yf, resid, b, h0 + 1, w, mbase, hi);
  }
}

// ---------------------------------------------------------------------------
// Zero the 1-px halo of a padded [8][66][66][IC] bf16 buffer (260 border px/b).
// ---------------------------------------------------------------------------
__global__ __launch_bounds__(256) void zero_halo(__bf16* __restrict__ X, int IC) {
  int idx = blockIdx.x * 256 + threadIdx.x;
  int total = 8 * 260 * IC;
  if (idx >= total) return;
  int c = idx % IC, r = idx / IC, p = r % 260, b = r / 260;
  int h, w;
  if (p < 66)       { h = 0;           w = p;        }
  else if (p < 132) { h = 65;          w = p - 66;   }
  else if (p < 196) { h = p - 132 + 1; w = 0;        }
  else              { h = p - 196 + 1; w = 65;       }
  X[((size_t)(b * PH + h) * PH + w) * IC + c] = (__bf16)0.0f;
}

// ---------------------------------------------------------------------------
// content [8,512,64,64] f32 NCHW -> padded [8,66,66,512] bf16 (LDS transpose)
// ---------------------------------------------------------------------------
__global__ __launch_bounds__(256) void pack_content(const float* __restrict__ content,
                                                    __bf16* __restrict__ Xt) {
  int bh = blockIdx.x;
  int b = bh >> 6, h = bh & 63;
  const float* src = content + (size_t)b * 512 * 4096 + h * 64;
  __bf16* dst = Xt + (((size_t)b * PH + (h + 1)) * PH + 1) * 512;
  __shared__ float tile[32][65];
  int t = threadIdx.x;
  for (int c0 = 0; c0 < 512; c0 += 32) {
    __syncthreads();
#pragma unroll
    for (int p = 0; p < 8; ++p) {
      int cl = p * 4 + (t >> 6), ww = t & 63;
      tile[cl][ww] = src[(size_t)(c0 + cl) * 4096 + ww];
    }
    __syncthreads();
#pragma unroll
    for (int p = 0; p < 8; ++p) {
      int ww = p * 8 + (t >> 5), cl = t & 31;
      dst[(size_t)ww * 512 + c0 + cl] = (__bf16)tile[cl][ww];
    }
  }
}

// ---------------------------------------------------------------------------
// Weight packing: A_ds [tap][32][512], A_up [tap][512][32] (bf16),
// Wp [c*9+tap][64] f32 (f1|f2 predictor conv weights for the pooled GEMV).
// ---------------------------------------------------------------------------
__global__ __launch_bounds__(256) void pack_weights(
    const float* __restrict__ ds_w, const float* __restrict__ up_w,
    const float* __restrict__ f1_cw, const float* __restrict__ f2_cw,
    __bf16* __restrict__ A_ds, __bf16* __restrict__ A_up, float* __restrict__ Wp) {
  int idx = blockIdx.x * 256 + threadIdx.x;
  if (blockIdx.z == 0) {
    if (idx < 9 * 32 * 512) {
      int c = idx % 512, r = idx / 512, oc = r % 32, tap = r / 32;
      A_ds[idx] = (__bf16)ds_w[(oc * 512 + c) * 9 + tap];
    }
  } else if (blockIdx.z == 1) {
    if (idx < 9 * 512 * 32) {
      int c = idx & 31, r = idx >> 5, oc = r % 512, tap = r / 512;
      A_up[idx] = (__bf16)up_w[(oc * 32 + c) * 9 + tap];
    }
  } else {
    if (idx < 4608 * 64) {
      int oc2 = idx & 63, k = idx >> 6, tap = k % 9, c = k / 9;
      Wp[idx] = (oc2 < 32) ? f1_cw[(oc2 * 512 + c) * 9 + tap]
                           : f2_cw[((oc2 - 32) * 512 + c) * 9 + tap];
    }
  }
}

// ---------------------------------------------------------------------------
// Per-(b,c) style edge statistics: S[b][c][tap] = sum over SAME-shifted window.
// mean(conv3x3(style)) then needs only S . W — kills both 9.7 GF style convs.
// ---------------------------------------------------------------------------
__global__ __launch_bounds__(256) void style_stats(const float* __restrict__ style,
                                                   float* __restrict__ S) {
  int bc = blockIdx.x;  // b*512 + c
  const float* x = style + (size_t)bc * 4096;
  int t = threadIdx.x;
  int hrow = t >> 2, w0 = (t & 3) * 16;
  float pT = 0.f, pr0 = 0.f, pr63 = 0.f, pc0 = 0.f, pc63 = 0.f;
#pragma unroll
  for (int i = 0; i < 16; ++i) {
    int ww = w0 + i;
    float v = x[hrow * 64 + ww];
    pT += v;
    if (hrow == 0)  pr0  += v;
    if (hrow == 63) pr63 += v;
    if (ww == 0)    pc0  += v;
    if (ww == 63)   pc63 += v;
  }
  __shared__ float red[5][256];
  red[0][t] = pT; red[1][t] = pr0; red[2][t] = pr63; red[3][t] = pc0; red[4][t] = pc63;
  __syncthreads();
  for (int s = 128; s > 0; s >>= 1) {
    if (t < s)
      for (int k2 = 0; k2 < 5; ++k2) red[k2][t] += red[k2][t + s];
    __syncthreads();
  }
  if (t == 0) {
    float T = red[0][0], r0 = red[1][0], r63 = red[2][0], cs0 = red[3][0], cs63 = red[4][0];
    float* So = S + (size_t)bc * 9;
#pragma unroll
    for (int tap = 0; tap < 9; ++tap) {
      int dy = tap / 3 - 1, dx = tap % 3 - 1;
      float s = T;
      if (dy == -1) s -= r63; else if (dy == 1) s -= r0;
      if (dx == -1) s -= cs63; else if (dx == 1) s -= cs0;
      if (dy != 0 && dx != 0) {
        int hr = (dy == -1) ? 63 : 0, wc = (dx == -1) ? 63 : 0;
        s += x[hr * 64 + wc];
      }
      So[tap] = s;
    }
  }
}

// pooled[b][oc2] = (1/4096) * sum_k S[b][k] * Wp[k][oc2] + cb[oc2]   (oc2: f1|f2)
__global__ void pooled_kernel(const float* __restrict__ S, const float* __restrict__ Wp,
                              const float* __restrict__ f1_cb, const float* __restrict__ f2_cb,
                              float* __restrict__ pooled) {
  int b = blockIdx.x, oc2 = threadIdx.x;  // 64 threads
  const float* Sb = S + (size_t)b * 4608;
  float acc = 0.f;
  for (int k = 0; k < 4608; ++k) acc += Sb[k] * Wp[(size_t)k * 64 + oc2];
  float bias = (oc2 < 32) ? f1_cb[oc2] : f2_cb[oc2 - 32];
  pooled[b * 64 + oc2] = acc * (1.0f / 4096.0f) + bias;
}

// FC (9216 x 32) per batch, output written directly as packed bf16 A-matrix
// Afilt[b][(tap*32+oc)*32+ic].
__global__ __launch_bounds__(256) void fc_pack(const float* __restrict__ pooled,
                                               const float* __restrict__ fw,
                                               const float* __restrict__ fb,
                                               __bf16* __restrict__ Afilt, int setOfs) {
  int j = blockIdx.x * 256 + threadIdx.x;
  int b = blockIdx.y;
  if (j >= 9216) return;
  const float* p = pooled + b * 64 + setOfs;
  const float* wr = fw + (size_t)j * 32;
  float acc = fb[j];
#pragma unroll 8
  for (int i = 0; i < 32; ++i) acc += p[i] * wr[i];
  int tap = j % 9, t2 = j / 9, ic = t2 & 31, oc = t2 >> 5;
  Afilt[(size_t)b * 9216 + (tap * 32 + oc) * 32 + ic] = (__bf16)acc;
}

// ---------------------------------------------------------------------------
extern "C" void kernel_launch(void* const* d_in, const int* in_sizes, int n_in,
                              void* d_out, int out_size, void* d_ws, size_t ws_size,
                              hipStream_t stream) {
  const float* content = (const float*)d_in[0];
  const float* style   = (const float*)d_in[1];
  const float* ds_w    = (const float*)d_in[2];
  const float* ds_b    = (const float*)d_in[3];
  const float* up_w    = (const float*)d_in[4];
  const float* up_b    = (const float*)d_in[5];
  const float* f1_cw   = (const float*)d_in[6];
  const float* f1_cb   = (const float*)d_in[7];
  const float* f1_fw   = (const float*)d_in[8];
  const float* f1_fb   = (const float*)d_in[9];
  const float* f2_cw   = (const float*)d_in[10];
  const float* f2_cb   = (const float*)d_in[11];
  const float* f2_fw   = (const float*)d_in[12];
  const float* f2_fb   = (const float*)d_in[13];

  char* ws = (char*)d_ws;
  size_t off = 0;
  auto take = [&](size_t bytes) {
    void* p = ws + off;
    off += (bytes + 255) & ~(size_t)255;
    return p;
  };
  __bf16* contentT = (__bf16*)take(8ull * PH * PH * 512 * 2);  // padded NHWC bf16
  __bf16* c_ds     = (__bf16*)take(8ull * PH * PH * 32 * 2);
  __bf16* m1       = (__bf16*)take(8ull * PH * PH * 32 * 2);
  __bf16* m2       = (__bf16*)take(8ull * PH * PH * 32 * 2);
  __bf16* A_ds     = (__bf16*)take(9ull * 32 * 512 * 2);
  __bf16* A_up     = (__bf16*)take(9ull * 512 * 32 * 2);
  __bf16* Af1      = (__bf16*)take(8ull * 9216 * 2);
  __bf16* Af2      = (__bf16*)take(8ull * 9216 * 2);
  float*  Wp       = (float*)take(4608ull * 64 * 4);
  float*  S        = (float*)take(8ull * 512 * 9 * 4);
  float*  pooled   = (float*)take(8ull * 64 * 4);
  (void)ws_size; (void)in_sizes; (void)n_in; (void)out_size;

  // Stage 0: halo zeroing, weight + activation packing, style statistics.
  zero_halo<<<(8 * 260 * 512 + 255) / 256, 256, 0, stream>>>(contentT, 512);
  zero_halo<<<(8 * 260 * 32 + 255) / 256, 256, 0, stream>>>(c_ds, 32);
  zero_halo<<<(8 * 260 * 32 + 255) / 256, 256, 0, stream>>>(m1, 32);
  zero_halo<<<(8 * 260 * 32 + 255) / 256, 256, 0, stream>>>(m2, 32);
  pack_weights<<<dim3(1152, 1, 3), 256, 0, stream>>>(ds_w, up_w, f1_cw, f2_cw,
                                                     A_ds, A_up, Wp);
  pack_content<<<512, 256, 0, stream>>>(content, contentT);
  style_stats<<<4096, 256, 0, stream>>>(style, S);

  // Stage 1: filter prediction (pooled means via edge stats, then FC -> bf16 A).
  pooled_kernel<<<8, 64, 0, stream>>>(S, Wp, f1_cb, f2_cb, pooled);
  fc_pack<<<dim3(36, 8), 256, 0, stream>>>(pooled, f1_fw, f1_fb, Af1, 0);
  fc_pack<<<dim3(36, 8), 256, 0, stream>>>(pooled, f2_fw, f2_fb, Af2, 32);

  // Stage 2: WMMA conv pipeline (2 output rows per wave).
  dim3 grid(32, 8);
  conv3x3_wmma<512, 32, false, false><<<grid, 256, 0, stream>>>(
      contentT, A_ds, 0, ds_b, c_ds, nullptr, nullptr);
  conv3x3_wmma<32, 32, true, false><<<grid, 256, 0, stream>>>(
      c_ds, Af1, 1, nullptr, m1, nullptr, nullptr);
  conv3x3_wmma<32, 32, false, false><<<grid, 256, 0, stream>>>(
      m1, Af2, 1, nullptr, m2, nullptr, nullptr);
  conv3x3_wmma<32, 512, false, true><<<grid, 256, 0, stream>>>(
      m2, A_up, 0, up_b, nullptr, (float*)d_out, content);
}
